// GeoEvidentialMappingLayer_90211493085316
// MI455X (gfx1250) — matrix operand order repaired and checked by reading, not compiled, not run on hardware
//
#include <hip/hip_runtime.h>
#include <hip/hip_bf16.h>
#include <stdint.h>

typedef __attribute__((ext_vector_type(16))) __bf16 bf16x16;
typedef __attribute__((ext_vector_type(8)))  __bf16 bf16x8;
typedef __attribute__((ext_vector_type(8)))  float  f32x8;
typedef __attribute__((ext_vector_type(2)))  int    v2i;
typedef __attribute__((ext_vector_type(4)))  int    v4i;

#define GEO_B  8
#define GEO_C  256
#define GEO_HW 16384
#define GEO_P  64

// ---- LDS layout (bytes) ----
#define CHUNK_RSTR   130                         // floats/row: 128 data + 2 pad (8B granule ok, breaks 64-bank aliasing)
#define CHUNK_BYTES  (64 * CHUNK_RSTR * 4)       // 33280
#define OFF_CHUNK0   0
#define OFF_CHUNK1   CHUNK_BYTES
#define CENT_RSTR_US 264                         // ushorts/row: 256 data + 8 pad (16B aligned rows)
#define OFF_CENT     (2 * CHUNK_BYTES)           // 66560
#define CENT_BYTES   (GEO_P * CENT_RSTR_US * 2)  // 33792
#define OFF_PAR      (OFF_CENT + CENT_BYTES)     // 100352 : psq|U0|U1|alphap|gam2 (5*64 f32)
#define OFF_XSQ      (OFF_PAR + 5 * 64 * 4)      // 101632 : per-lane |x|^2 partials (256 f32)
#define SMEM_TOTAL   (OFF_XSQ + 256 * 4)         // 102656

// ---- workspace layout (bytes) ----
#define WS_CENT 0
#define WS_PSQ  32768
#define WS_U0   33024
#define WS_U1   33280
#define WS_AP   33536
#define WS_G2   33792

// ---------------- CDNA5 async global->LDS helpers ----------------
#if __has_builtin(__builtin_amdgcn_global_load_async_to_lds_b64) && \
    __has_builtin(__builtin_amdgcn_global_load_async_to_lds_b128)
#define ASYNC_VIA_BUILTIN 1
#else
#define ASYNC_VIA_BUILTIN 0
#endif

typedef __attribute__((address_space(1))) void* as1p;
typedef __attribute__((address_space(3))) void* as3p;
typedef __attribute__((address_space(1))) v2i*  g_v2i_p;
typedef __attribute__((address_space(3))) v2i*  l_v2i_p;
typedef __attribute__((address_space(1))) v4i*  g_v4i_p;
typedef __attribute__((address_space(3))) v4i*  l_v4i_p;

__device__ __forceinline__ void async_ld_b64(const void* g, void* l) {
#if ASYNC_VIA_BUILTIN
  __builtin_amdgcn_global_load_async_to_lds_b64((g_v2i_p)(as1p)g, (l_v2i_p)(as3p)l, 0, 0);
#else
  unsigned            lofs = (unsigned)(uintptr_t)(as3p)l;
  unsigned long long  ga   = (unsigned long long)(uintptr_t)g;
  asm volatile("global_load_async_to_lds_b64 %0, %1, off" :: "v"(lofs), "v"(ga) : "memory");
#endif
}

__device__ __forceinline__ void async_ld_b128(const void* g, void* l) {
#if ASYNC_VIA_BUILTIN
  __builtin_amdgcn_global_load_async_to_lds_b128((g_v4i_p)(as1p)g, (l_v4i_p)(as3p)l, 0, 0);
#else
  unsigned            lofs = (unsigned)(uintptr_t)(as3p)l;
  unsigned long long  ga   = (unsigned long long)(uintptr_t)g;
  asm volatile("global_load_async_to_lds_b128 %0, %1, off" :: "v"(lofs), "v"(ga) : "memory");
#endif
}

__device__ __forceinline__ void wait_async0() {
#if __has_builtin(__builtin_amdgcn_s_wait_asynccnt)
  __builtin_amdgcn_s_wait_asynccnt(0);
#else
  asm volatile("s_wait_asynccnt 0" ::: "memory");
#endif
}

// ---------------- precompute: scale, bf16 centers, p_sq, U, alphap, gam2 ----------------
__global__ void geo_precompute(const float* __restrict__ cm, const float* __restrict__ alpha,
                               const float* __restrict__ gamma, const float* __restrict__ pc,
                               const float* __restrict__ gc,
                               unsigned short* __restrict__ centb, float* __restrict__ psq,
                               float* __restrict__ u0, float* __restrict__ u1,
                               float* __restrict__ ap, float* __restrict__ g2) {
  __shared__ float s_scale;
  const int p = threadIdx.x;  // 64 threads
  if (p == 0) {
    float s = 0.f;
    for (int i = 0; i < GEO_P; ++i) s += gc[i];
    s_scale = 1.0f + 0.1f * tanhf(s * (1.0f / GEO_P));
  }
  __syncthreads();
  const float scale = s_scale;

  ap[p] = 0.99f / (1.0f + __expf(-alpha[p]));
  const float g = gamma[p];
  g2[p] = g * g;
  const float c0 = cm[2 * p], c1 = cm[2 * p + 1];
  const float b0 = c0 * c0, b1 = c1 * c1;
  const float inv = 1.0f / (b0 + b1);
  u0[p] = b0 * inv;
  u1[p] = b1 * inv;

  float ps = 0.f;
  for (int c = 0; c < GEO_C; ++c) {
    const float v = pc[p * GEO_C + c] * scale;
    ps += v * v;
    const __bf16 h = (__bf16)v;
    centb[p * GEO_C + c] = __builtin_bit_cast(unsigned short, h);
  }
  psq[p] = ps;
}

// ---------------- main fused kernel ----------------
__global__ __launch_bounds__(256, 1) void geo_main(
    const float* __restrict__ feats, const unsigned short* __restrict__ centb,
    const float* __restrict__ psq, const float* __restrict__ u0,
    const float* __restrict__ u1, const float* __restrict__ apv,
    const float* __restrict__ g2v, float* __restrict__ out) {
  extern __shared__ char smem[];
  const int tid  = threadIdx.x;
  const int wave = tid >> 5;
  const int lane = tid & 31;
  const int col  = lane & 15;                 // pixel column within wave tile
  const int b    = blockIdx.x >> 7;
  const int pixb = (blockIdx.x & 127) * 128;  // 128 contiguous pixels of image b
  const float* featb = feats + (size_t)b * GEO_C * GEO_HW;

  // per-prototype params -> LDS
  float* par = (float*)(smem + OFF_PAR);
  if (tid < 64) {
    par[tid]       = psq[tid];
    par[64 + tid]  = u0[tid];
    par[128 + tid] = u1[tid];
    par[192 + tid] = apv[tid];
    par[256 + tid] = g2v[tid];
  }

  // async-copy bf16 centers (64 rows x 512B) into padded LDS (528B rows)
  {
    const char* g = (const char*)centb;
#pragma unroll
    for (int it = 0; it < 8; ++it) {
      const int row = wave + it * 8;
      async_ld_b128(g + row * 512 + lane * 16, smem + OFF_CENT + row * 528 + lane * 16);
    }
  }

  // async prefetch of a 64ch x 128px f32 chunk into padded LDS rows (520B stride)
  auto prefetch = [&](int k) {
    char* lbase = smem + ((k & 1) ? OFF_CHUNK1 : OFF_CHUNK0);
#pragma unroll
    for (int i = 0; i < 16; ++i) {
      const int gidx = tid + i * 256;
      const int row  = gidx >> 6;
      const int c8   = gidx & 63;
      async_ld_b64((const char*)featb + ((size_t)(k * 64 + row) * GEO_HW + pixb) * 4 + c8 * 8,
                   lbase + row * (CHUNK_RSTR * 4) + c8 * 8);
    }
  };
  prefetch(0);

  f32x8 acc[4];
#pragma unroll
  for (int t = 0; t < 4; ++t)
#pragma unroll
    for (int e = 0; e < 8; ++e) acc[t][e] = 0.f;

  float xs_acc = 0.f;
  const int hi16 = (lane >= 16) ? 16 : 0;  // B: lanes 16-31 carry K=16..31
  const int hi8  = (lane >= 16) ? 8 : 0;   // C/D: lanes 16-31 hold rows 8..15
  const unsigned short* cent = (const unsigned short*)(smem + OFF_CENT);

  for (int k = 0; k < 4; ++k) {
    wait_async0();
    __syncthreads();                // chunk k (and, at k==0, centers) visible to all
    if (k < 3) prefetch(k + 1);     // overlap next chunk with compute
    const float* ch = (const float*)(smem + ((k & 1) ? OFF_CHUNK1 : OFF_CHUNK0));
#pragma unroll
    for (int kk = 0; kk < 2; ++kk) {
      // B fragment: this lane's pixel, 16 consecutive channels; fuse |x|^2 partial
      bf16x16 bfr;
      float xs = 0.f;
#pragma unroll
      for (int j = 0; j < 16; ++j) {
        const float f = ch[(kk * 32 + hi16 + j) * CHUNK_RSTR + wave * 16 + col];
        xs += f * f;
        bfr[j] = (__bf16)f;
      }
      xs_acc += xs;
      const int cA = k * 64 + kk * 32 + hi16;  // A K-base for this lane (K = cA..cA+15)
#pragma unroll
      for (int t = 0; t < 4; ++t) {
        const unsigned short* rowp = cent + (t * 16 + col) * CENT_RSTR_US;
        const bf16x8 lo = *(const bf16x8*)(rowp + cA);        // K = cA..cA+7
        const bf16x8 hi = *(const bf16x8*)(rowp + cA + 8);    // K = cA+8..cA+15
        bf16x16 afr;
#pragma unroll
        for (int e = 0; e < 8; ++e) { afr[e] = lo[e]; afr[e + 8] = hi[e]; }
        acc[t] = __builtin_amdgcn_wmma_f32_16x16x32_bf16(false, afr, false, bfr, (short)0,
                                                         acc[t], false, false);
      }
    }
    __syncthreads();                // everyone done reading this buffer before reuse
  }

  // stash x.p tiles + |x|^2 partials (reuse chunk buffer 0: last chunk was buffer 1)
  float* xsqp = (float*)(smem + OFF_XSQ);
  xsqp[wave * 32 + lane] = xs_acc;
  float* sbuf = (float*)(smem + OFF_CHUNK0) + wave * 1024;  // 64 protos x 16 px
#pragma unroll
  for (int t = 0; t < 4; ++t)
#pragma unroll
    for (int r = 0; r < 8; ++r) sbuf[(t * 16 + r + hi8) * 16 + col] = acc[t][r];
  __syncthreads();

  // sequential evidential combine: one pixel per lane (lanes 0-15 of each wave)
  if (lane < 16) {
    const float xsq = xsqp[wave * 32 + lane] + xsqp[wave * 32 + lane + 16];
    float ms0 = 0.f, ms1 = 0.f, mt = 1.f;
    for (int p = 0; p < GEO_P; ++p) {
      const float xdot = sbuf[p * 16 + lane];
      const float d    = 0.5f * xsq - xdot + 0.5f * par[p];
      const float s    = par[192 + p] * __expf(-par[256 + p] * d);
      const float m0   = s * par[64 + p];
      const float m1   = s * par[128 + p];
      const float mth  = 1.0f - s;
      ms0 = ms0 * (m0 + mth) + m0 * mt;
      ms1 = ms1 * (m1 + mth) + m1 * mt;
      mt *= mth;
    }
    const float inv = 1.0f / (ms0 + ms1 + mt + 1e-12f);
    const int hw = pixb + wave * 16 + lane;
    float* ob = out + (size_t)b * 3 * GEO_HW;
    ob[hw]              = ms0 * inv;
    ob[GEO_HW + hw]     = ms1 * inv;
    ob[2 * GEO_HW + hw] = mt * inv;
  }
}

extern "C" void kernel_launch(void* const* d_in, const int* in_sizes, int n_in,
                              void* d_out, int out_size, void* d_ws, size_t ws_size,
                              hipStream_t stream) {
  (void)in_sizes; (void)n_in; (void)out_size; (void)ws_size;
  const float* feats = (const float*)d_in[0];
  // d_in[1] = geo_context (unused by the reference math)
  const float* cm    = (const float*)d_in[2];
  const float* alpha = (const float*)d_in[3];
  const float* gamma = (const float*)d_in[4];
  const float* pc    = (const float*)d_in[5];
  const float* gc    = (const float*)d_in[6];

  char* ws = (char*)d_ws;
  unsigned short* centb = (unsigned short*)(ws + WS_CENT);
  float* psq = (float*)(ws + WS_PSQ);
  float* u0  = (float*)(ws + WS_U0);
  float* u1  = (float*)(ws + WS_U1);
  float* ap  = (float*)(ws + WS_AP);
  float* g2  = (float*)(ws + WS_G2);

  geo_precompute<<<1, 64, 0, stream>>>(cm, alpha, gamma, pc, gc, centb, psq, u0, u1, ap, g2);
  geo_main<<<dim3(GEO_B * (GEO_HW / 128)), dim3(256), SMEM_TOTAL, stream>>>(
      feats, centb, psq, u0, u1, ap, g2, (float*)d_out);
}
